// RelativeMSAAttention_48326972015123
// MI455X (gfx1250) — matrix operand order repaired and checked by reading, not compiled, not run on hardware
//
#include <hip/hip_runtime.h>
#include <hip/hip_bf16.h>

// ---------------------------------------------------------------------------
// Types for WMMA bf16: D(f32 16x16) = A(bf16 16x32) * B(bf16 32x16) + C
// ---------------------------------------------------------------------------
typedef __attribute__((ext_vector_type(16))) __bf16 v16bf;
typedef __attribute__((ext_vector_type(8)))  float  v8f;
typedef __attribute__((ext_vector_type(4)))  unsigned int v4u;
typedef __attribute__((ext_vector_type(8)))  int    v8i;
typedef __attribute__((ext_vector_type(4)))  int    v4i;

union Frag16 { v16bf v; unsigned int u[8]; };

__device__ __forceinline__ unsigned short f32_to_bf16(float f) {
    union { float f; unsigned int u; } c; c.f = f;
    unsigned int u = c.u;
    unsigned int r = (u + 0x7FFFu + ((u >> 16) & 1u)) >> 16;   // RNE
    return (unsigned short)r;
}

// A-matrix (16x32 bf16), source row-major [M][K], lda in elements.
// contiguous k order used here (same as B): lanes0-15 M=0..15 K=0..15; lanes16-31 K=16..31
__device__ __forceinline__ void load_a_rowmajor(Frag16& A, const unsigned short* base,
                                                int lda, int m, int hi) {
    const unsigned short* row = base + m * lda;
#pragma unroll
    for (int j = 0; j < 8; ++j) {
        int k = hi * 16 + 2 * j;
        A.u[j] = *(const unsigned int*)(row + k);
    }
}

// B-matrix (32x16 bf16) where B[k][n] = src[n*ldn + k]  (each n-row K-contiguous)
// contiguous k order used here (same as A): lanes0-15 K=0..15, lanes16-31 K=16..31, N = lane%16
__device__ __forceinline__ void load_b_kcontig(Frag16& B, const unsigned short* base,
                                               int ldn, int n, int hi) {
    const unsigned short* row = base + n * ldn;
#pragma unroll
    for (int j = 0; j < 8; ++j) {
        int k = hi * 16 + 2 * j;
        B.u[j] = *(const unsigned int*)(row + k);
    }
}

// B-matrix (32x16 bf16) where B[k][n] = src[k*ldk + n]  (K along rows)
__device__ __forceinline__ void load_b_strided(Frag16& B, const unsigned short* base,
                                               int ldk, int n, int hi) {
#pragma unroll
    for (int j = 0; j < 8; ++j) {
        int k = hi * 16 + 2 * j;
        unsigned int lo  = base[(k    ) * ldk + n];
        unsigned int hi16= base[(k + 1) * ldk + n];
        B.u[j] = lo | (hi16 << 16);
    }
}

__device__ __forceinline__ v8f wmma_bf16(const Frag16& A, const Frag16& B, v8f C) {
    return __builtin_amdgcn_wmma_f32_16x16x32_bf16(false, A.v, false, B.v,
                                                   (short)0, C, false, false);
}

// ---------------------------------------------------------------------------
// Problem constants
// ---------------------------------------------------------------------------
#define MSA 64
#define LSEQ 256
#define DMODEL 512
#define NHEAD 8
#define DH 64
#define HM (NHEAD * MSA)          // 512
#define NROW (MSA * LSEQ)          // 16384 rows of x
#define LL (LSEQ * LSEQ)           // 65536

// ---------------------------------------------------------------------------
// 1) f32 -> bf16 convert
// ---------------------------------------------------------------------------
__global__ void cvt_bf16_kernel(const float* __restrict__ src,
                                unsigned short* __restrict__ dst, int n) {
    int t = blockIdx.x * blockDim.x + threadIdx.x;
    if (t < n) dst[t] = f32_to_bf16(src[t]);
}

// ---------------------------------------------------------------------------
// head-split store of a 16x16 f32 tile as bf16 q/k/v
// ---------------------------------------------------------------------------
__device__ __forceinline__ void store_head_split(unsigned short* outp, int rowTile,
                                                 int e, int hi, const v8f& c) {
    const int h = e >> 6, d = e & 63;
#pragma unroll
    for (int r = 0; r < 8; ++r) {
        int rowg = rowTile * 16 + r + hi * 8;
        int m = rowg >> 8, i = rowg & 255;
        outp[((size_t)((h * MSA + m) * LSEQ + i)) * DH + d] = f32_to_bf16(c[r]);
    }
}

// ---------------------------------------------------------------------------
// 2) QKV projection: y[row,e] = sum_d x[row,d] * W[e,d], head-split bf16 out
//    one wave = one 16x64 strip (A-frag reused across 4 B tiles)
//    waves = (NROW/16) * (3*DMODEL/64) = 1024*24 = 24576 -> 3072 blocks
// ---------------------------------------------------------------------------
__global__ void __launch_bounds__(256)
qkv_proj_kernel(const unsigned short* __restrict__ xb,
                const unsigned short* __restrict__ wb,
                unsigned short* __restrict__ qb,
                unsigned short* __restrict__ kb,
                unsigned short* __restrict__ vb) {
    const int lane = threadIdx.x & 31;
    const int wave = (blockIdx.x * blockDim.x + threadIdx.x) >> 5;
    const int n16 = lane & 15, hi = lane >> 4;

    const int rowTile = wave / 24;
    const int colGrp  = wave % 24;
    const int eb    = colGrp * 64;       // 64-wide, never straddles q/k/v
    const int which = eb >> 9;           // 0=q 1=k 2=v
    const int el0   = eb & 511;

    const unsigned short* W    = wb + which * (DMODEL * DMODEL);
    const unsigned short* Arow = xb + (size_t)rowTile * 16 * DMODEL;

    v8f c0 = {}, c1 = {}, c2 = {}, c3 = {};
    for (int kb0 = 0; kb0 < DMODEL; kb0 += 32) {
        Frag16 A, B;
        load_a_rowmajor(A, Arow + kb0, DMODEL, n16, hi);
        load_b_kcontig(B, W + (el0 +  0) * DMODEL + kb0, DMODEL, n16, hi);
        c0 = wmma_bf16(A, B, c0);
        load_b_kcontig(B, W + (el0 + 16) * DMODEL + kb0, DMODEL, n16, hi);
        c1 = wmma_bf16(A, B, c1);
        load_b_kcontig(B, W + (el0 + 32) * DMODEL + kb0, DMODEL, n16, hi);
        c2 = wmma_bf16(A, B, c2);
        load_b_kcontig(B, W + (el0 + 48) * DMODEL + kb0, DMODEL, n16, hi);
        c3 = wmma_bf16(A, B, c3);
    }

    unsigned short* outp = (which == 0) ? qb : (which == 1) ? kb : vb;
    store_head_split(outp, rowTile, el0 +  0 + n16, hi, c0);
    store_head_split(outp, rowTile, el0 + 16 + n16, hi, c1);
    store_head_split(outp, rowTile, el0 + 32 + n16, hi, c2);
    store_head_split(outp, rowTile, el0 + 48 + n16, hi, c3);
}

// ---------------------------------------------------------------------------
// 3) e1: S[hm][i][j] = 0.125 * sum_d q[hm][i][d] k[hm][j][d]
//    One block per hm. TDM stages q[hm] (32KB) -> LDS[0] and k[hm] (32KB) ->
//    LDS[32768]; 8 waves then compute all 256 16x16 tiles out of LDS.
// ---------------------------------------------------------------------------
__global__ void __launch_bounds__(256)
e1_kernel(const unsigned short* __restrict__ qb,
          const unsigned short* __restrict__ kb,
          float* __restrict__ S) {
    __shared__ __align__(16) unsigned short smem[32768];   // 64 KB: q | k
    const int hm = blockIdx.x;
    const int lane = threadIdx.x & 31;
    const int waveId = threadIdx.x >> 5;
    const int n16 = lane & 15, hi = lane >> 4;

#if __has_builtin(__builtin_amdgcn_tensor_load_to_lds) && __has_builtin(__builtin_amdgcn_s_wait_tensorcnt)
    if (waveId == 0) {
        unsigned long long qa = (unsigned long long)(qb + (size_t)hm * (LSEQ * DH));
        unsigned long long ka = (unsigned long long)(kb + (size_t)hm * (LSEQ * DH));
        // D# group0: count=1 | lds_addr | global_addr | type=2
        v4u g0q = { 1u, 0u, (unsigned)(qa & 0xffffffffu),
                    (unsigned)((qa >> 32) & 0x01ffffffu) | (2u << 30) };
        v4u g0k = { 1u, 32768u, (unsigned)(ka & 0xffffffffu),
                    (unsigned)((ka >> 32) & 0x01ffffffu) | (2u << 30) };
        // D# group1: 2D tensor, data_size=2B, dim0=64, dim1=256, tile=64x256,
        //            dim0_stride=64 elements
        v8i g1 = { (int)(1u << 16),     // workgroup_mask=0, data_size=1 (2B)
                   (int)(64u << 16),    // tensor_dim0[15:0] in bits [63:48]
                   (int)(256u << 16),   // tensor_dim0 hi=0 | tensor_dim1[15:0]
                   (int)(64u << 16),    // tensor_dim1 hi=0 | tile_dim0=64
                   (int)(256u),         // tile_dim1=256, tile_dim2=0
                   (int)(64u),          // tensor_dim0_stride lo = 64
                   0, 0 };
        v4i gz4 = { 0, 0, 0, 0 };
        v8i gz8 = { 0, 0, 0, 0, 0, 0, 0, 0 };
        __builtin_amdgcn_tensor_load_to_lds(g0q, g1, gz4, gz4, gz8, 0);
        __builtin_amdgcn_tensor_load_to_lds(g0k, g1, gz4, gz4, gz8, 0);
        __builtin_amdgcn_s_wait_tensorcnt(0);
    }
#else
    {   // cooperative copy fallback
        const uint4* gq = (const uint4*)(qb + (size_t)hm * (LSEQ * DH));
        const uint4* gk = (const uint4*)(kb + (size_t)hm * (LSEQ * DH));
        uint4* sq = (uint4*)(&smem[0]);
        uint4* sk = (uint4*)(&smem[16384]);
        for (int t = threadIdx.x; t < 2048; t += 256) { sq[t] = gq[t]; sk[t] = gk[t]; }
    }
#endif
    __syncthreads();

    // 8 waves x 2 i-tiles each; A-frags (from LDS) reused across all 16 j-tiles
#pragma unroll
    for (int ii = 0; ii < 2; ++ii) {
        const int iT = waveId * 2 + ii;
        Frag16 A0, A1;
        load_a_rowmajor(A0, smem + iT * 16 * DH + 0,  DH, n16, hi);
        load_a_rowmajor(A1, smem + iT * 16 * DH + 32, DH, n16, hi);
        for (int jT = 0; jT < 16; ++jT) {
            Frag16 B0, B1;
            load_b_kcontig(B0, smem + 16384 + jT * 16 * DH + 0,  DH, n16, hi);
            load_b_kcontig(B1, smem + 16384 + jT * 16 * DH + 32, DH, n16, hi);
            v8f c = {};
            c = wmma_bf16(A0, B0, c);
            c = wmma_bf16(A1, B1, c);
            float* Sp = S + (size_t)hm * LL + jT * 16 + n16;
#pragma unroll
            for (int r = 0; r < 8; ++r) {
                int i = iT * 16 + r + hi * 8;
                Sp[(size_t)i * LSEQ] = c[r] * 0.125f;
            }
        }
    }
}

// ---------------------------------------------------------------------------
// 4) e2 (batched over i): S[hm][i][j] += 0.125 * sum_d q[hm][i][d] aK[i][j][d]
//    one wave = 16 hm x 64 j (A-frag reused over 4 j tiles)
//    waves = LSEQ * (HM/16) * (LSEQ/64) = 256*32*4 = 32768 -> 4096 blocks
// ---------------------------------------------------------------------------
__global__ void __launch_bounds__(256)
e2_kernel(const unsigned short* __restrict__ qb,
          const unsigned short* __restrict__ aKb,
          float* __restrict__ S) {
    const int lane = threadIdx.x & 31;
    const int wave = (blockIdx.x * blockDim.x + threadIdx.x) >> 5;
    const int n16 = lane & 15, hi = lane >> 4;

    const int i   = wave >> 7;
    const int rem = wave & 127;
    const int hmT = rem >> 2, jG = rem & 3;          // jG selects 4 j-tiles

    const unsigned short* Ab = qb + (size_t)(hmT * 16) * (LSEQ * DH) + i * DH; // lda=LSEQ*DH
    const unsigned short* Bb = aKb + (size_t)i * (LSEQ * DH);

    v8f c[4] = {{}, {}, {}, {}};
#pragma unroll
    for (int kb0 = 0; kb0 < DH; kb0 += 32) {
        Frag16 A, B;
        load_a_rowmajor(A, Ab + kb0, LSEQ * DH, n16, hi);
#pragma unroll
        for (int t = 0; t < 4; ++t) {
            int jT = jG * 4 + t;
            load_b_kcontig(B, Bb + jT * 16 * DH + kb0, DH, n16, hi);
            c[t] = wmma_bf16(A, B, c[t]);
        }
    }

#pragma unroll
    for (int t = 0; t < 4; ++t) {
        int jT = jG * 4 + t;
        float* Sp = S + (size_t)i * LSEQ + jT * 16 + n16;
#pragma unroll
        for (int r = 0; r < 8; ++r) {
            int hm = hmT * 16 + r + hi * 8;
            float* p = Sp + (size_t)hm * LL;
            *p += c[t][r] * 0.125f;
        }
    }
}

// ---------------------------------------------------------------------------
// 5) softmax over j per (hm,i) row; one wave per row; alpha -> bf16 P
// ---------------------------------------------------------------------------
__global__ void softmax_kernel(const float* __restrict__ S,
                               unsigned short* __restrict__ P) {
    const int lane = threadIdx.x & 31;
    const int row = (blockIdx.x * blockDim.x + threadIdx.x) >> 5;  // HM*LSEQ rows
    const float* sp = S + (size_t)row * LSEQ;

    float vals[8];
    float mx = -3.4e38f;
#pragma unroll
    for (int t = 0; t < 8; ++t) { vals[t] = sp[lane + t * 32]; mx = fmaxf(mx, vals[t]); }
#pragma unroll
    for (int o = 16; o > 0; o >>= 1) mx = fmaxf(mx, __shfl_xor(mx, o, 32));

    float sum = 0.f;
#pragma unroll
    for (int t = 0; t < 8; ++t) { vals[t] = __expf(vals[t] - mx); sum += vals[t]; }
#pragma unroll
    for (int o = 16; o > 0; o >>= 1) sum += __shfl_xor(sum, o, 32);

    float inv = 1.0f / sum;
    unsigned short* pp = P + (size_t)row * LSEQ;
#pragma unroll
    for (int t = 0; t < 8; ++t) pp[lane + t * 32] = f32_to_bf16(vals[t] * inv);
}

// ---------------------------------------------------------------------------
// 6) z1: Z[hm][i][d] = sum_j P[hm][i][j] v[hm][j][d]
//    one wave = 16 i x 64 d (full DH strip, A-frag reused over 4 d tiles)
//    waves = HM * (LSEQ/16) = 8192 -> 1024 blocks
// ---------------------------------------------------------------------------
__global__ void __launch_bounds__(256)
z1_kernel(const unsigned short* __restrict__ P,
          const unsigned short* __restrict__ vb,
          float* __restrict__ Z) {
    const int lane = threadIdx.x & 31;
    const int wave = (blockIdx.x * blockDim.x + threadIdx.x) >> 5;
    const int n16 = lane & 15, hi = lane >> 4;

    const int hm = wave >> 4;
    const int iT = wave & 15;

    const unsigned short* Ab = P  + (size_t)hm * LL + iT * 16 * LSEQ;
    const unsigned short* Bb = vb + (size_t)hm * (LSEQ * DH);

    v8f c[4] = {{}, {}, {}, {}};
    for (int kb0 = 0; kb0 < LSEQ; kb0 += 32) {
        Frag16 A, B;
        load_a_rowmajor(A, Ab + kb0, LSEQ, n16, hi);
#pragma unroll
        for (int t = 0; t < 4; ++t) {
            load_b_strided(B, Bb + (size_t)kb0 * DH + t * 16, DH, n16, hi);
            c[t] = wmma_bf16(A, B, c[t]);
        }
    }

#pragma unroll
    for (int t = 0; t < 4; ++t) {
        float* Zp = Z + (size_t)hm * (LSEQ * DH) + t * 16 + n16;
#pragma unroll
        for (int r = 0; r < 8; ++r) {
            int i = iT * 16 + r + hi * 8;
            Zp[(size_t)i * DH] = c[t][r];
        }
    }
}

// ---------------------------------------------------------------------------
// 7) z2 (batched over i): Z[hm][i][d] += sum_j P[hm][i][j] aV[i][j][d]
//    one wave = 16 hm x 64 d; waves = LSEQ * (HM/16) = 8192 -> 1024 blocks
// ---------------------------------------------------------------------------
__global__ void __launch_bounds__(256)
z2_kernel(const unsigned short* __restrict__ P,
          const unsigned short* __restrict__ aVb,
          float* __restrict__ Z) {
    const int lane = threadIdx.x & 31;
    const int wave = (blockIdx.x * blockDim.x + threadIdx.x) >> 5;
    const int n16 = lane & 15, hi = lane >> 4;

    const int i   = wave >> 5;
    const int hmT = wave & 31;

    const unsigned short* Ab = P   + (size_t)(hmT * 16) * LL + (size_t)i * LSEQ; // lda = LL
    const unsigned short* Bb = aVb + (size_t)i * (LSEQ * DH);

    v8f c[4] = {{}, {}, {}, {}};
    for (int kb0 = 0; kb0 < LSEQ; kb0 += 32) {
        Frag16 A, B;
        load_a_rowmajor(A, Ab + kb0, LL, n16, hi);
#pragma unroll
        for (int t = 0; t < 4; ++t) {
            load_b_strided(B, Bb + (size_t)kb0 * DH + t * 16, DH, n16, hi);
            c[t] = wmma_bf16(A, B, c[t]);
        }
    }

#pragma unroll
    for (int t = 0; t < 4; ++t) {
        float* Zp = Z + (size_t)i * DH + t * 16 + n16;
#pragma unroll
        for (int r = 0; r < 8; ++r) {
            int hm = hmT * 16 + r + hi * 8;
            Zp[(size_t)hm * (LSEQ * DH)] += c[t][r];
        }
    }
}

// ---------------------------------------------------------------------------
// 8) reduce over MSA: out[i*512 + h*64 + d] = sum_m Z[(h*64+m)][i][d]
// ---------------------------------------------------------------------------
__global__ void reduce_kernel(const float* __restrict__ Z, float* __restrict__ out) {
    int t = blockIdx.x * blockDim.x + threadIdx.x;     // 131072
    if (t >= LSEQ * DMODEL) return;
    int d = t & 63;
    int h = (t >> 6) & 7;
    int i = t >> 9;
    float s = 0.f;
    for (int m = 0; m < MSA; ++m)
        s += Z[((size_t)((h * MSA + m) * LSEQ + i)) * DH + d];
    out[t] = s;
}

// ---------------------------------------------------------------------------
// Host launcher
// ---------------------------------------------------------------------------
extern "C" void kernel_launch(void* const* d_in, const int* in_sizes, int n_in,
                              void* d_out, int out_size, void* d_ws, size_t ws_size,
                              hipStream_t stream) {
    (void)in_sizes; (void)n_in; (void)out_size; (void)ws_size;

    const float* x  = (const float*)d_in[0];
    const float* Wq = (const float*)d_in[1];
    const float* Wk = (const float*)d_in[2];
    const float* Wv = (const float*)d_in[3];
    const float* aK = (const float*)d_in[4];
    const float* aV = (const float*)d_in[5];
    float* out = (float*)d_out;

    char* ws = (char*)d_ws;
    size_t o = 0;
    unsigned short* xb  = (unsigned short*)(ws + o); o += (size_t)NROW * DMODEL * 2;        // 16.8 MB
    unsigned short* wb  = (unsigned short*)(ws + o); o += (size_t)3 * DMODEL * DMODEL * 2;  // 1.6 MB
    unsigned short* qb  = (unsigned short*)(ws + o); o += (size_t)HM * LSEQ * DH * 2;       // 16.8 MB
    unsigned short* kb  = (unsigned short*)(ws + o); o += (size_t)HM * LSEQ * DH * 2;
    unsigned short* vb  = (unsigned short*)(ws + o); o += (size_t)HM * LSEQ * DH * 2;
    unsigned short* aKb = (unsigned short*)(ws + o); o += (size_t)LSEQ * LSEQ * DH * 2;     // 8.4 MB
    unsigned short* aVb = (unsigned short*)(ws + o); o += (size_t)LSEQ * LSEQ * DH * 2;
    float*          S   = (float*)(ws + o);         o += (size_t)HM * LL * 4;               // 134 MB
    unsigned short* P   = (unsigned short*)(ws + o); o += (size_t)HM * LL * 2;              // 67 MB
    float*          Z   = (float*)(ws + o);         o += (size_t)HM * LSEQ * DH * 4;        // 33.5 MB

    const int T = 256;
    // 1) converts
    cvt_bf16_kernel<<<(NROW * DMODEL + T - 1) / T, T, 0, stream>>>(x,  xb,  NROW * DMODEL);
    cvt_bf16_kernel<<<(DMODEL * DMODEL + T - 1) / T, T, 0, stream>>>(Wq, wb,                       DMODEL * DMODEL);
    cvt_bf16_kernel<<<(DMODEL * DMODEL + T - 1) / T, T, 0, stream>>>(Wk, wb + DMODEL * DMODEL,     DMODEL * DMODEL);
    cvt_bf16_kernel<<<(DMODEL * DMODEL + T - 1) / T, T, 0, stream>>>(Wv, wb + 2 * DMODEL * DMODEL, DMODEL * DMODEL);
    cvt_bf16_kernel<<<(LSEQ * LSEQ * DH + T - 1) / T, T, 0, stream>>>(aK, aKb, LSEQ * LSEQ * DH);
    cvt_bf16_kernel<<<(LSEQ * LSEQ * DH + T - 1) / T, T, 0, stream>>>(aV, aVb, LSEQ * LSEQ * DH);

    // 2) QKV projection: 24576 waves -> 3072 blocks
    qkv_proj_kernel<<<3072, T, 0, stream>>>(xb, wb, qb, kb, vb);

    // 3) e1: one block per hm (TDM-staged LDS)
    e1_kernel<<<HM, T, 0, stream>>>(qb, kb, S);
    // 4) e2 accumulate: 32768 waves -> 4096 blocks
    e2_kernel<<<4096, T, 0, stream>>>(qb, aKb, S);
    // 5) softmax: 131072 rows, one wave each
    softmax_kernel<<<16384, T, 0, stream>>>(S, P);
    // 6) z1: 8192 waves -> 1024 blocks
    z1_kernel<<<1024, T, 0, stream>>>(P, vb, Z);
    // 7) z2 accumulate: 8192 waves -> 1024 blocks
    z2_kernel<<<1024, T, 0, stream>>>(P, aVb, Z);
    // 8) reduce over MSA
    reduce_kernel<<<(LSEQ * DMODEL + T - 1) / T, T, 0, stream>>>(Z, out);
}
